// MultiHeadedAttention_9646496546931
// MI455X (gfx1250) — matrix operand order repaired and checked by reading, not compiled, not run on hardware
//
#include <hip/hip_runtime.h>
#include <hip/hip_bf16.h>
#include <math.h>

#define B_BATCH 2
#define S_LEN   2048
#define D_MODEL 1024
#define H_HEADS 16
#define DK      64
#define KT      64    // key-tile for flash attention

typedef __attribute__((ext_vector_type(16))) __bf16        v16bf;
typedef __attribute__((ext_vector_type(8)))  float         v8f;
typedef __attribute__((ext_vector_type(4)))  unsigned int  u32x4;
typedef __attribute__((ext_vector_type(4)))  int           i32x4;
typedef __attribute__((ext_vector_type(4)))  float         f32x4;

union FragA { v16bf v; u32x4 q[2]; unsigned short s[16]; };
union FragC { v8f   v; float f[8]; };

#ifndef __has_builtin
#define __has_builtin(x) 0
#endif

#if __has_builtin(__builtin_amdgcn_global_load_async_to_lds_b128)
#define ASYNC_LDS 1
#else
#define ASYNC_LDS 0
#endif

#if __has_builtin(__builtin_amdgcn_s_wait_asynccnt)
#define WAIT_ASYNC() __builtin_amdgcn_s_wait_asynccnt(0)
#elif ASYNC_LDS
#define WAIT_ASYNC() asm volatile("s_wait_asynccnt 0x0" ::: "memory")
#else
#define WAIT_ASYNC() ((void)0)
#endif

__device__ __forceinline__ unsigned short f2bf(float x) {
    unsigned int u = __float_as_uint(x);
    u += 0x7FFFu + ((u >> 16) & 1u);   // round-to-nearest-even
    return (unsigned short)(u >> 16);
}
__device__ __forceinline__ unsigned int pack2(float a, float b) {
    return (unsigned int)f2bf(a) | ((unsigned int)f2bf(b) << 16);
}

// 16-byte global -> LDS copy: async DMA on CDNA5, sync fallback otherwise.
__device__ __forceinline__ void copy16(const unsigned short* g, unsigned short* l) {
#if ASYNC_LDS
    __builtin_amdgcn_global_load_async_to_lds_b128(
        (__attribute__((address_space(1))) i32x4*)g,
        (__attribute__((address_space(3))) i32x4*)l, 0, 0);
#else
    *(u32x4*)l = *(const u32x4*)g;
#endif
}

// ---------------------------------------------------------------------------
// Tiled GEMM:  Out(M,N) = A(M,K) @ W(N,K)^T + bias(N)
//   A_BF16      : A operand is bf16 (ctx) vs f32 (activations)
//   MODE 0      : store bf16, head-split  [B][H][S][DK]   (Q, K)
//   MODE 1      : store bf16, head-split transposed [B][H][DK][S]  (V)
//   MODE 2      : store f32 row-major (final output)
// Block: 256 threads (8 waves), tile 128(M) x 128(N) x 32(K).
// ---------------------------------------------------------------------------
template <bool A_BF16, int MODE>
__global__ __launch_bounds__(256) void gemm_wmma_bf16(
    const void* __restrict__ Aptr, const float* __restrict__ W,
    const float* __restrict__ bias, void* __restrict__ Out,
    int M, int N, int K)
{
    __shared__ __align__(16) unsigned short lA[128 * 32];
    __shared__ __align__(16) unsigned short lB[128 * 32];

    const int tid  = threadIdx.x;
    const int wave = tid >> 5;
    const int lane = tid & 31;
    const int h    = lane >> 4;
    const int r    = lane & 15;
    const int wm   = wave >> 2;
    const int wn   = wave & 3;
    const int blockM = blockIdx.y * 128;
    const int blockN = blockIdx.x * 128;

    FragC acc[4][2];
#pragma unroll
    for (int i = 0; i < 4; ++i)
#pragma unroll
        for (int j = 0; j < 2; ++j)
#pragma unroll
            for (int e = 0; e < 8; ++e) acc[i][j].f[e] = 0.0f;

    const int loadRow = tid >> 1;
    const int loadCol = (tid & 1) * 16;

    for (int k0 = 0; k0 < K; k0 += 32) {
        if (A_BF16) {
            const unsigned short* A16 = (const unsigned short*)Aptr;
            const u32x4* src = (const u32x4*)(A16 + (size_t)(blockM + loadRow) * K + k0 + loadCol);
            *(u32x4*)&lA[loadRow * 32 + loadCol]     = src[0];
            *(u32x4*)&lA[loadRow * 32 + loadCol + 8] = src[1];
        } else {
            const float* A32 = (const float*)Aptr;
            const f32x4* src = (const f32x4*)(A32 + (size_t)(blockM + loadRow) * K + k0 + loadCol);
            f32x4 a0 = src[0], a1 = src[1], a2 = src[2], a3 = src[3];
            u32x4 o0, o1;
            o0[0] = pack2(a0[0], a0[1]); o0[1] = pack2(a0[2], a0[3]);
            o0[2] = pack2(a1[0], a1[1]); o0[3] = pack2(a1[2], a1[3]);
            o1[0] = pack2(a2[0], a2[1]); o1[1] = pack2(a2[2], a2[3]);
            o1[2] = pack2(a3[0], a3[1]); o1[3] = pack2(a3[2], a3[3]);
            *(u32x4*)&lA[loadRow * 32 + loadCol]     = o0;
            *(u32x4*)&lA[loadRow * 32 + loadCol + 8] = o1;
        }
        {
            const f32x4* src = (const f32x4*)(W + (size_t)(blockN + loadRow) * K + k0 + loadCol);
            f32x4 a0 = src[0], a1 = src[1], a2 = src[2], a3 = src[3];
            u32x4 o0, o1;
            o0[0] = pack2(a0[0], a0[1]); o0[1] = pack2(a0[2], a0[3]);
            o0[2] = pack2(a1[0], a1[1]); o0[3] = pack2(a1[2], a1[3]);
            o1[0] = pack2(a2[0], a2[1]); o1[1] = pack2(a2[2], a2[3]);
            o1[2] = pack2(a3[0], a3[1]); o1[3] = pack2(a3[2], a3[3]);
            *(u32x4*)&lB[loadRow * 32 + loadCol]     = o0;
            *(u32x4*)&lB[loadRow * 32 + loadCol + 8] = o1;
        }
        __syncthreads();

        FragA af[4];
#pragma unroll
        for (int i = 0; i < 4; ++i) {
            int row = wm * 64 + i * 16 + r;
            af[i].q[0] = *(const u32x4*)&lA[row * 32 + 8 * h];
            af[i].q[1] = *(const u32x4*)&lA[row * 32 + 16 + 8 * h];
        }
        FragA bf[2];
#pragma unroll
        for (int j = 0; j < 2; ++j) {
            int col = wn * 32 + j * 16 + r;
            bf[j].q[0] = *(const u32x4*)&lB[col * 32 + 8 * h];
            bf[j].q[1] = *(const u32x4*)&lB[col * 32 + 16 + 8 * h];
        }
#pragma unroll
        for (int i = 0; i < 4; ++i)
#pragma unroll
            for (int j = 0; j < 2; ++j)
                acc[i][j].v = __builtin_amdgcn_wmma_f32_16x16x32_bf16(
                    false, af[i].v, false, bf[j].v, (short)0, acc[i][j].v, false, false);
        __syncthreads();
    }

#pragma unroll
    for (int i = 0; i < 4; ++i)
#pragma unroll
        for (int j = 0; j < 2; ++j) {
            int rowBase = blockM + wm * 64 + i * 16 + 8 * h;
            int col     = blockN + wn * 32 + j * 16 + r;
            float bv = bias[col];
#pragma unroll
            for (int e = 0; e < 8; ++e) {
                int   row = rowBase + e;
                float val = acc[i][j].f[e] + bv;
                if (MODE == 0) {
                    unsigned short* O16 = (unsigned short*)Out;
                    int b = row / S_LEN, s = row % S_LEN;
                    int head = col / DK, dk = col % DK;
                    O16[(((size_t)b * H_HEADS + head) * S_LEN + s) * DK + dk] = f2bf(val);
                } else if (MODE == 1) {
                    unsigned short* O16 = (unsigned short*)Out;
                    int b = row / S_LEN, s = row % S_LEN;
                    int head = col / DK, dk = col % DK;
                    O16[(((size_t)b * H_HEADS + head) * DK + dk) * S_LEN + s] = f2bf(val);
                } else {
                    ((float*)Out)[(size_t)row * N + col] = val;
                }
            }
        }
}

// ---------------------------------------------------------------------------
// Flash attention. Grid: (S/128, B*H). 8 waves; wave w owns 16 query rows.
// 64-key tiles, double-buffered async global->LDS staging.
// Q/K workspace layout [B][H][S][DK]; V workspace TRANSPOSED [B][H][DK][S]
// so both K and V tiles are straight strided copies (async-friendly) and all
// WMMA fragment fetches are contiguous ds_load_b128.
// ---------------------------------------------------------------------------
__global__ __launch_bounds__(256) void flash_attn_wmma(
    const unsigned short* __restrict__ Q, const unsigned short* __restrict__ Kd,
    const unsigned short* __restrict__ Vt, const float* __restrict__ posr,
    const int* __restrict__ mask, unsigned short* __restrict__ Ctx)
{
    __shared__ __align__(16) unsigned short lK[2][KT * DK];   // [key][dk]
    __shared__ __align__(16) unsigned short lV[2][DK * KT];   // [dk][key]
    __shared__ __align__(16) unsigned short lP[8][16 * KT];

    const int tid  = threadIdx.x;
    const int wave = tid >> 5;
    const int lane = tid & 31;
    const int h    = lane >> 4;
    const int r    = lane & 15;
    const int bh   = blockIdx.y;
    const int b    = bh / H_HEADS;
    const int head = bh % H_HEADS;
    const int q0   = blockIdx.x * 128 + wave * 16;

    const size_t baseQK = (size_t)bh * S_LEN * DK;   // Q,K: [bh][S][DK]
    const size_t baseV  = (size_t)bh * DK * S_LEN;   // V:   [bh][DK][S]

    // Q fragments (rows q0+r), direct A-layout loads from global
    FragA qf[2];
#pragma unroll
    for (int f = 0; f < 2; ++f) {
        const unsigned short* qp = Q + baseQK + (size_t)(q0 + r) * DK + f * 32;
        qf[f].q[0] = *(const u32x4*)(qp + 8 * h);
        qf[f].q[1] = *(const u32x4*)(qp + 16 + 8 * h);
    }

    float m_i[8], l_i[8];
    FragC o[4];
#pragma unroll
    for (int e = 0; e < 8; ++e) { m_i[e] = -1e30f; l_i[e] = 0.0f; }
#pragma unroll
    for (int t = 0; t < 4; ++t)
#pragma unroll
        for (int e = 0; e < 8; ++e) o[t].f[e] = 0.0f;

    const float scale = 0.125f;  // 1/sqrt(64)

    // stage one KT x DK K-tile and DK x KT V-tile into buffer p
    auto stage = [&](int j, int p) {
#pragma unroll
        for (int i = 0; i < 2; ++i) {
            int c   = tid + 256 * i;          // 0..511
            int row = c >> 3;                 // K: key row / V: dk row (0..63)
            int c8  = (c & 7) * 8;            // 8-short chunk within row
            copy16(Kd + baseQK + (size_t)(j + row) * DK + c8, &lK[p][row * DK + c8]);
            copy16(Vt + baseV + (size_t)row * S_LEN + j + c8, &lV[p][row * KT + c8]);
        }
    };

    stage(0, 0);

    for (int j = 0; j < S_LEN; j += KT) {
        const int p = (j >> 6) & 1;
        WAIT_ASYNC();            // our async copies for tile j have landed
        __syncthreads();         // everyone's copies visible; prev buffer free
        if (j + KT < S_LEN) stage(j + KT, p ^ 1);

        // prefetch next tile's posr rows (read-once stream)
        if (j + KT < S_LEN && r == 0) {
#pragma unroll
            for (int e = 0; e < 8; ++e)
                __builtin_prefetch(posr + ((size_t)b * S_LEN + q0 + 8 * h + e) * S_LEN + j + KT, 0, 1);
        }

        // ---- scores: 16 q-rows x 64 keys (8 WMMAs) ----
        FragC sc[4];
#pragma unroll
        for (int kb = 0; kb < 4; ++kb) {
#pragma unroll
            for (int e = 0; e < 8; ++e) sc[kb].f[e] = 0.0f;
#pragma unroll
            for (int f = 0; f < 2; ++f) {
                FragA kf;
                const unsigned short* kp = &lK[p][(kb * 16 + r) * DK + f * 32 + 8 * h];
                kf.q[0] = *(const u32x4*)kp;
                kf.q[1] = *(const u32x4*)(kp + 16);
                sc[kb].v = __builtin_amdgcn_wmma_f32_16x16x32_bf16(
                    false, qf[f].v, false, kf.v, (short)0, sc[kb].v, false, false);
            }
        }

        int mk[4];
#pragma unroll
        for (int kb = 0; kb < 4; ++kb) mk[kb] = mask[b * S_LEN + j + kb * 16 + r];

        // ---- posr + mask + online softmax ----
#pragma unroll
        for (int e = 0; e < 8; ++e) {
            int qrow = q0 + 8 * h + e;
            const float* pr = posr + ((size_t)b * S_LEN + qrow) * S_LEN + j;
            float s[4];
#pragma unroll
            for (int kb = 0; kb < 4; ++kb) {
                s[kb] = sc[kb].f[e] * scale + pr[kb * 16 + r];
                if (mk[kb] == 0) s[kb] = -1e9f;
            }
            float mx = fmaxf(fmaxf(s[0], s[1]), fmaxf(s[2], s[3]));
#pragma unroll
            for (int off = 1; off < 16; off <<= 1) mx = fmaxf(mx, __shfl_xor(mx, off, 16));
            float mnew = fmaxf(m_i[e], mx);
            float corr = __expf(m_i[e] - mnew);
            m_i[e] = mnew;

            float pv[4], ps = 0.0f;
#pragma unroll
            for (int kb = 0; kb < 4; ++kb) { pv[kb] = __expf(s[kb] - mnew); ps += pv[kb]; }
#pragma unroll
            for (int off = 1; off < 16; off <<= 1) ps += __shfl_xor(ps, off, 16);
            l_i[e] = l_i[e] * corr + ps;
#pragma unroll
            for (int t = 0; t < 4; ++t) o[t].f[e] *= corr;
#pragma unroll
            for (int kb = 0; kb < 4; ++kb)
                lP[wave][(8 * h + e) * KT + kb * 16 + r] = f2bf(pv[kb]);
        }

        // ---- P (A-layout, 16x64) and O += P V (8 WMMAs) ----
        FragA pf[2];
#pragma unroll
        for (int f = 0; f < 2; ++f) {
            pf[f].q[0] = *(const u32x4*)&lP[wave][r * KT + f * 32 + 8 * h];
            pf[f].q[1] = *(const u32x4*)&lP[wave][r * KT + f * 32 + 16 + 8 * h];
        }
#pragma unroll
        for (int t = 0; t < 4; ++t) {
#pragma unroll
            for (int f = 0; f < 2; ++f) {
                FragA vf;
                const unsigned short* vp = &lV[p][(t * 16 + r) * KT + f * 32 + 8 * h];
                vf.q[0] = *(const u32x4*)vp;
                vf.q[1] = *(const u32x4*)(vp + 16);
                o[t].v = __builtin_amdgcn_wmma_f32_16x16x32_bf16(
                    false, pf[f].v, false, vf.v, (short)0, o[t].v, false, false);
            }
        }
    }

    // ---- normalize + store ctx (merged heads, bf16) ----
    float inv[8];
#pragma unroll
    for (int e = 0; e < 8; ++e) inv[e] = (l_i[e] > 0.0f) ? (1.0f / l_i[e]) : 0.0f;
#pragma unroll
    for (int t = 0; t < 4; ++t)
#pragma unroll
        for (int e = 0; e < 8; ++e) {
            int row = q0 + 8 * h + e;
            int col = head * DK + t * 16 + r;
            Ctx[((size_t)b * S_LEN + row) * D_MODEL + col] = f2bf(o[t].f[e] * inv[e]);
        }
}

// ---------------------------------------------------------------------------
extern "C" void kernel_launch(void* const* d_in, const int* in_sizes, int n_in,
                              void* d_out, int out_size, void* d_ws, size_t ws_size,
                              hipStream_t stream) {
    (void)in_sizes; (void)n_in; (void)out_size; (void)ws_size;
    const float* query = (const float*)d_in[0];
    const float* key   = (const float*)d_in[1];
    const float* value = (const float*)d_in[2];
    const int*   mask  = (const int*)d_in[3];
    const float* posr  = (const float*)d_in[4];
    const float* Wq = (const float*)d_in[5];  const float* bq = (const float*)d_in[6];
    const float* Wk = (const float*)d_in[7];  const float* bk = (const float*)d_in[8];
    const float* Wv = (const float*)d_in[9];  const float* bv = (const float*)d_in[10];
    const float* Wo = (const float*)d_in[11]; const float* bo = (const float*)d_in[12];
    float* out = (float*)d_out;

    const size_t nElem = (size_t)B_BATCH * S_LEN * D_MODEL;
    unsigned short* qws = (unsigned short*)d_ws;
    unsigned short* kws = qws + nElem;
    unsigned short* vws = kws + nElem;   // transposed [B][H][DK][S]
    unsigned short* cws = vws + nElem;

    const int M = B_BATCH * S_LEN;           // 4096
    dim3 gGemm(D_MODEL / 128, M / 128);      // (8, 32)
    dim3 blk(256);

    gemm_wmma_bf16<false, 0><<<gGemm, blk, 0, stream>>>(query, Wq, bq, qws, M, D_MODEL, D_MODEL);
    gemm_wmma_bf16<false, 0><<<gGemm, blk, 0, stream>>>(key,   Wk, bk, kws, M, D_MODEL, D_MODEL);
    gemm_wmma_bf16<false, 1><<<gGemm, blk, 0, stream>>>(value, Wv, bv, vws, M, D_MODEL, D_MODEL);

    dim3 gFlash(S_LEN / 128, B_BATCH * H_HEADS);  // (16, 32)
    flash_attn_wmma<<<gFlash, blk, 0, stream>>>(qws, kws, vws, posr, mask, cws);

    gemm_wmma_bf16<true, 2><<<gGemm, blk, 0, stream>>>(cws, Wo, bo, out, M, D_MODEL, D_MODEL);
}
